// MultiBranchDH_SFNN_2637109920468
// MI455X (gfx1250) — matrix operand order repaired and checked by reading, not compiled, not run on hardware
//
#include <hip/hip_runtime.h>
#include <hip/hip_bf16.h>

// Problem constants (from the reference)
#define B_   256
#define T_   2048
#define IN_  128
#define H_   256
#define NB_  4
#define BS_  64
#define OUT_ 8
#define TT_  16   // timesteps processed per WMMA tile iteration

typedef __attribute__((ext_vector_type(16))) __bf16 v16bf;
typedef __attribute__((ext_vector_type(8)))  float  v8f;

union Frag {
  v16bf          v;
  unsigned short s[16];
  uint4          q[2];
};

__device__ __forceinline__ unsigned bf16_bits(float f) {
  union { float f; unsigned u; } cv; cv.f = f;
  unsigned u = cv.u;
  u += 0x7FFFu + ((u >> 16) & 1u);   // round-to-nearest-even
  return (u >> 16);
}

__device__ __forceinline__ unsigned pack_bf16x2(float lo, float hi) {
  return bf16_bits(lo) | (bf16_bits(hi) << 16);
}

__device__ __forceinline__ float sigmoidf_(float xv) {
  return 1.0f / (1.0f + __expf(-xv));
}

// One block per batch row b. 256 threads = 8 wave32.
// Fused: bf16-WMMA branch GEMM -> register-resident EMA scan -> bf16-WMMA
// output projection -> sigmoid -> store. Recurrent state never leaves VGPRs.
__global__ __launch_bounds__(256, 1)
void sfnn_fused_kernel(const float* __restrict__ x,      // [B,T,IN]
                       const float* __restrict__ Wb,     // [NB,IN,BS]
                       const float* __restrict__ bb,     // [NB,BS] -> flat [H]
                       const float* __restrict__ Wo,     // [H,OUT]
                       const float* __restrict__ bo,     // [OUT]
                       const float* __restrict__ tau_m,  // [H]
                       const float* __restrict__ tau_n,  // [NB,BS] -> flat [H]
                       float* __restrict__ out)          // [B,T,OUT]
{
  __shared__ __align__(16) unsigned short sX[TT_ * IN_];    // 4 KB  x tile (bf16)
  __shared__ __align__(16) float          sD[TT_ * H_];     // 16 KB dendritic tile
  __shared__ __align__(16) unsigned short sM[TT_ * H_];     // 8 KB  membrane tile (bf16)
  __shared__ __align__(16) float          sP[8 * TT_ * 16]; // 8 KB  per-wave output partials

  const int tid  = threadIdx.x;          // == channel h for the scan phase
  const int b    = blockIdx.x;
  const int w    = tid >> 5;             // wave id 0..7
  const int lane = tid & 31;
  const int half = lane >> 4;            // 0 | 1 (lane group)
  const int l16  = lane & 15;

  // ------------------------------------------------------------------
  // One-time: B-matrix fragments for the branch GEMM, held in VGPRs.
  // Wave w owns H-column tiles ht0 = 2w, ht1 = 2w+1 (16 columns each).
  // B layout (32x16 bf16): N = lane%16; K = 16*(lane/16) + j  (j = elem 0..15)
  // Combined weight W[i][h] = Wb[h/64][i][h%64].
  // ------------------------------------------------------------------
  Frag wf[2][4];
  #pragma unroll
  for (int tile = 0; tile < 2; ++tile) {
    const int hcol = (2 * w + tile) * 16 + l16;       // N -> h column
    const int n    = hcol >> 6;
    const int s    = hcol & 63;
    #pragma unroll
    for (int kk = 0; kk < 4; ++kk) {
      Frag u;
      #pragma unroll
      for (int j = 0; j < 16; ++j) {
        const int i = kk * 32 + (half ? 16 + j : j);  // K -> input dim
        u.s[j] = (unsigned short)bf16_bits(Wb[(n * IN_ + i) * BS_ + s]);
      }
      wf[tile][kk] = u;
    }
  }

  // One-time: Wo fragment (output projection, K-split across waves).
  // Wave w covers K(h) = w*32 .. w*32+31; N = output col (8..15 zero-padded).
  Frag wof;
  {
    #pragma unroll
    for (int j = 0; j < 16; ++j) {
      const int hh = w * 32 + (half ? 16 + j : j);
      const float v = (l16 < OUT_) ? Wo[hh * OUT_ + l16] : 0.0f;
      wof.s[j] = (unsigned short)bf16_bits(v);
    }
  }

  // Per-channel scan constants + recurrent state (registers, whole T loop).
  const float alpha = sigmoidf_(tau_m[tid]);
  const float beta  = sigmoidf_(tau_n[tid]);   // tau_n flat == reshape(-1)
  const float bbh   = bb[tid];
  const float bo_o  = (tid < TT_ * OUT_) ? bo[tid & 7] : 0.0f;
  float c = 0.0f, m = 0.0f;

  const float* xrow = x + (size_t)b * T_ * IN_;
  unsigned* sX32 = (unsigned*)sX;

  for (int t0 = 0; t0 < T_; t0 += TT_) {
    const float* xp = xrow + (size_t)t0 * IN_;

    // ---- stage x tile [16 x 128] into LDS as packed bf16 ----
    // 512 float4 global loads (b128, fully coalesced), 2 per thread,
    // packed to 2x b32 LDS stores each.
    {
      const float4* xp4 = (const float4*)xp;
      #pragma unroll
      for (int k = 0; k < 2; ++k) {
        const int q = tid + k * 256;
        const float4 v = xp4[q];
        sX32[q * 2 + 0] = pack_bf16x2(v.x, v.y);
        sX32[q * 2 + 1] = pack_bf16x2(v.z, v.w);
      }
    }
    // Prefetch next tile into caches (global_prefetch_b8): 8 KB / 256 thr.
    if (t0 + TT_ < T_) {
      __builtin_prefetch((const char*)(xp + TT_ * IN_) + tid * 32, 0, 0);
    }
    __syncthreads();

    // ---- branch GEMM: D[16 x H] = X[16 x 128] @ W[128 x H] (bf16 WMMA) ----
    // A layout (16x32 bf16): M = lane%16 (= t); lanes 0-15 cover K
    // {0..15}, lanes 16-31 cover {16..31}.
    // Two 16-byte LDS reads per fragment.
    v8f acc0 = {}; v8f acc1 = {};
    #pragma unroll
    for (int kk = 0; kk < 4; ++kk) {
      Frag a;
      const unsigned short* base = &sX[l16 * IN_ + kk * 32 + half * 16];
      a.q[0] = *(const uint4*)(base);
      a.q[1] = *(const uint4*)(base + 8);
      acc0 = __builtin_amdgcn_wmma_f32_16x16x32_bf16(
                 false, a.v, false, wf[0][kk].v, (short)0, acc0, false, false);
      acc1 = __builtin_amdgcn_wmma_f32_16x16x32_bf16(
                 false, a.v, false, wf[1][kk].v, (short)0, acc1, false, false);
    }

    // C/D layout: N = lane%16, M = r + 8*(lane/16)
    #pragma unroll
    for (int r = 0; r < 8; ++r) {
      const int trow = r + 8 * half;
      sD[trow * H_ + (2 * w)     * 16 + l16] = acc0[r];
      sD[trow * H_ + (2 * w + 1) * 16 + l16] = acc1[r];
    }
    __syncthreads();

    // ---- sequential EMA scan: thread == channel h, 16 steps ----
    #pragma unroll
    for (int t = 0; t < TT_; ++t) {
      const float d = sD[t * H_ + tid] + bbh;
      c = beta  * c + (1.0f - beta)  * d;
      m = alpha * m + (1.0f - alpha) * c;
      sM[t * H_ + tid] = (unsigned short)bf16_bits(m);
    }
    __syncthreads();

    // ---- output projection: [16 x 256] @ [256 x 16pad], K split over waves ----
    {
      Frag a;
      const unsigned short* base = &sM[l16 * H_ + w * 32 + half * 16];
      a.q[0] = *(const uint4*)(base);
      a.q[1] = *(const uint4*)(base + 8);
      v8f po = {};
      po = __builtin_amdgcn_wmma_f32_16x16x32_bf16(
               false, a.v, false, wof.v, (short)0, po, false, false);
      #pragma unroll
      for (int r = 0; r < 8; ++r) {
        sP[w * (TT_ * 16) + (r + 8 * half) * 16 + l16] = po[r];
      }
    }
    __syncthreads();

    // ---- reduce 8 wave-partials, sigmoid, store [16 x 8] outputs ----
    if (tid < TT_ * OUT_) {
      const int t = tid >> 3;
      const int o = tid & 7;
      float ssum = 0.0f;
      #pragma unroll
      for (int wv = 0; wv < 8; ++wv) ssum += sP[wv * (TT_ * 16) + t * 16 + o];
      out[((size_t)b * T_ + t0 + t) * OUT_ + o] = sigmoidf_(ssum + bo_o);
    }
    __syncthreads();
  }
}

extern "C" void kernel_launch(void* const* d_in, const int* in_sizes, int n_in,
                              void* d_out, int out_size, void* d_ws, size_t ws_size,
                              hipStream_t stream) {
  const float* x     = (const float*)d_in[0];
  const float* Wb    = (const float*)d_in[1];
  const float* bb    = (const float*)d_in[2];
  const float* Wo    = (const float*)d_in[3];
  const float* bo    = (const float*)d_in[4];
  const float* tau_m = (const float*)d_in[5];
  const float* tau_n = (const float*)d_in[6];
  float* out = (float*)d_out;

  sfnn_fused_kernel<<<dim3(B_), dim3(256), 0, stream>>>(
      x, Wb, bb, Wo, bo, tau_m, tau_n, out);
}
